// MultiHeadSelfAttention_28475633172487
// MI455X (gfx1250) — compile-verified, matrix operand order checked by reading, not run on hardware
//
#include <hip/hip_runtime.h>

// ---------------------------------------------------------------------------
// Multi-head self-attention for MI455X (gfx1250, wave32, WMMA).
// Compute-bound (AI ~1500 FLOP/byte >> machine balance) -> f16 WMMA, f32 acc.
// Pipeline: cast -> QKV GEMMs (WMMA, async double-buffered LDS) ->
//           flash attention (WMMA scores + WMMA row-sums + DPP row-max) ->
//           out GEMM (fp32 result).
// ---------------------------------------------------------------------------

#define EMBED   1024
#define NHEADS  16
#define HDIM    64
#define BATCH   4
#define SEQ     2048
#define MTOT    (BATCH * SEQ)   // 8192

typedef _Float16 f16;
typedef _Float16 v16h __attribute__((ext_vector_type(16)));
typedef float    v8f  __attribute__((ext_vector_type(8)));

union FragU {
    v16h  v;
    uint4 q[2];
    f16   e[16];
};

// ---------------- async global->LDS path (CDNA5, ASYNCcnt) -----------------
#if defined(__has_builtin)
#if __has_builtin(__builtin_amdgcn_global_load_async_to_lds_b128) && \
    __has_builtin(__builtin_amdgcn_s_wait_asynccnt)
#define USE_ASYNC 1
#endif
#endif
#ifndef USE_ASYNC
#define USE_ASYNC 0
#endif

#if USE_ASYNC
// Builtin parameter types (from hipcc diagnostic): pointee is int4 vector,
// global source in AS1 (prints as "__device__"), LDS dest in AS3.
typedef int i32x4 __attribute__((vector_size(16)));
typedef __attribute__((address_space(1))) i32x4 gls_t;
typedef __attribute__((address_space(3))) i32x4 lds_t;
__device__ __forceinline__ void async_cp16(f16* dst, const f16* src) {
    // 16 bytes per lane, memory -> LDS directly (no VGPR round trip)
    __builtin_amdgcn_global_load_async_to_lds_b128(
        (gls_t*)(uintptr_t)src,
        (lds_t*)(uint32_t)(uintptr_t)dst, 0, 0);
}
__device__ __forceinline__ void async_wait0() {
    __builtin_amdgcn_s_wait_asynccnt(0);
}
#endif

__device__ __forceinline__ v8f zero8() {
    v8f z = {0.f, 0.f, 0.f, 0.f, 0.f, 0.f, 0.f, 0.f};
    return z;
}

__device__ __forceinline__ v8f wmma_f16(v16h a, v16h b, v8f c) {
    // emits v_wmma_f32_16x16x32_f16
    return __builtin_amdgcn_wmma_f32_16x16x32_f16(
        false, a, false, b, (short)0, c, false, false);
}

// All-lanes max within each 16-lane half via DPP row rotations (VALU only,
// co-executes with XDL WMMA; the C-matrix halves are exactly DPP rows).
__device__ __forceinline__ float rowmax16(float x) {
    int v = __builtin_bit_cast(int, x);
    x = fmaxf(x, __builtin_bit_cast(float, __builtin_amdgcn_update_dpp(v, v, 0x121, 0xf, 0xf, false)));
    v = __builtin_bit_cast(int, x);
    x = fmaxf(x, __builtin_bit_cast(float, __builtin_amdgcn_update_dpp(v, v, 0x122, 0xf, 0xf, false)));
    v = __builtin_bit_cast(int, x);
    x = fmaxf(x, __builtin_bit_cast(float, __builtin_amdgcn_update_dpp(v, v, 0x124, 0xf, 0xf, false)));
    v = __builtin_bit_cast(int, x);
    x = fmaxf(x, __builtin_bit_cast(float, __builtin_amdgcn_update_dpp(v, v, 0x128, 0xf, 0xf, false)));
    return x;
}

// Load a 16x32 (rows x K) fragment whose K dimension is contiguous in memory.
// Lane l<16: row (rowbase+l), K={0..7,16..23}; lane l+16: same row, K={8..15,24..31}.
__device__ __forceinline__ v16h load_frag_kcontig(const f16* tile, int rowbase,
                                                  int stride, int kcol, int lane) {
    const int r  = rowbase + (lane & 15);
    const int kh = (lane >> 4) << 3;  // 0 or 8
    const f16* p = tile + (size_t)r * stride + kcol + kh;
    FragU f;
    f.q[0] = *(const uint4*)(p);       // K = kh .. kh+7
    f.q[1] = *(const uint4*)(p + 16);  // K = 16+kh .. 16+kh+7
    return f.v;
}

// ---------------------------------------------------------------------------
// fp32 -> f16 cast (streaming)
// ---------------------------------------------------------------------------
__global__ void cast_f32_f16(const float* __restrict__ src, f16* __restrict__ dst, int n) {
    int i      = blockIdx.x * blockDim.x + threadIdx.x;
    int stride = gridDim.x * blockDim.x;
    for (; i < n; i += stride) dst[i] = (f16)src[i];
}

// ---------------------------------------------------------------------------
// GEMM: C[m,n] = sum_k A[m,k] * Wt[n,k] + bias[n]
// MODE 1: out f16 permuted to [B, H, S, D]  (QKV projections)
// MODE 2: out f32, [M, N]                    (output projection)
// 256 thr (8 waves), 128x128x32 tiles, waves 4(M)x2(N), 2x4 WMMA acc each.
// Double-buffered LDS; slabs streamed with async global->LDS when available.
// ---------------------------------------------------------------------------
template <int MODE>
__global__ void __launch_bounds__(256)
gemm_wmma(const f16* __restrict__ A, const f16* __restrict__ Wt,
          const float* __restrict__ bias, void* __restrict__ out,
          int M, int N, int K) {
    __shared__ f16 As[2][128 * 32];
    __shared__ f16 Bs[2][128 * 32];

    const int tid  = threadIdx.x;
    const int lane = tid & 31;
    const int wave = tid >> 5;
    const int wm   = wave & 3;
    const int wn   = wave >> 2;
    const int m0   = blockIdx.x * 128;
    const int n0   = blockIdx.y * 128;

    v8f acc[2][4];
#pragma unroll
    for (int i = 0; i < 2; ++i)
#pragma unroll
        for (int j = 0; j < 4; ++j) acc[i][j] = zero8();

    const int srow = tid >> 1;
    const int scol = (tid & 1) << 4;

    auto stage = [&](int bufi, int k0) {
        const f16* ga = A  + (size_t)(m0 + srow) * K + k0 + scol;
        const f16* gb = Wt + (size_t)(n0 + srow) * K + k0 + scol;
        f16* la = &As[bufi][srow * 32 + scol];
        f16* lb = &Bs[bufi][srow * 32 + scol];
#if USE_ASYNC
        async_cp16(la,     ga);
        async_cp16(la + 8, ga + 8);
        async_cp16(lb,     gb);
        async_cp16(lb + 8, gb + 8);
#else
        *(uint4*)(la)     = *(const uint4*)(ga);
        *(uint4*)(la + 8) = *(const uint4*)(ga + 8);
        *(uint4*)(lb)     = *(const uint4*)(gb);
        *(uint4*)(lb + 8) = *(const uint4*)(gb + 8);
        if (k0 + 32 < K) {
            __builtin_prefetch(ga + 32, 0, 1);
            __builtin_prefetch(gb + 32, 0, 1);
        }
#endif
    };

    stage(0, 0);
    int buf = 0;
    for (int k0 = 0; k0 < K; k0 += 32) {
#if USE_ASYNC
        async_wait0();
#endif
        __syncthreads();
        if (k0 + 32 < K) stage(buf ^ 1, k0 + 32);  // overlap next slab with compute

        v16h af[2], bf[4];
#pragma unroll
        for (int i = 0; i < 2; ++i)
            af[i] = load_frag_kcontig(As[buf], wm * 32 + i * 16, 32, 0, lane);
#pragma unroll
        for (int j = 0; j < 4; ++j)
            bf[j] = load_frag_kcontig(Bs[buf], wn * 64 + j * 16, 32, 0, lane);
#pragma unroll
        for (int i = 0; i < 2; ++i)
#pragma unroll
            for (int j = 0; j < 4; ++j)
                acc[i][j] = wmma_f16(af[i], bf[j], acc[i][j]);
        buf ^= 1;
    }

    // Epilogue. C layout: VGPR r, lanes 0-15 -> (M=r, N=lane); lanes 16-31 -> (M=r+8).
    const int colL  = lane & 15;
    const int rhalf = (lane >> 4) << 3;
#pragma unroll
    for (int i = 0; i < 2; ++i) {
#pragma unroll
        for (int j = 0; j < 4; ++j) {
            const int nn = n0 + wn * 64 + j * 16 + colL;
            const float bv = bias[nn];
#pragma unroll
            for (int r = 0; r < 8; ++r) {
                const int mm = m0 + wm * 32 + i * 16 + r + rhalf;
                const float val = acc[i][j][r] + bv;
                if (MODE == 1) {
                    const int h = nn >> 6, d = nn & 63;
                    const int b = mm >> 11, s = mm & (SEQ - 1);
                    ((f16*)out)[(((size_t)(b * NHEADS + h)) * SEQ + s) * HDIM + d] = (f16)val;
                } else {
                    ((float*)out)[(size_t)mm * N + nn] = val;
                }
            }
        }
    }
}

// ---------------------------------------------------------------------------
// Flash attention: grid (S/128, B*H), block 256 (8 waves).
// Wave owns 16 query rows (Q resident in VGPRs). Block streams 32-key tiles:
// K staged row-major (async), V staged TRANSPOSED ([d][key]) so PV
// B-fragments are contiguous-K b128 loads. Row max via DPP rotations (VALU);
// row sums via an extra WMMA against an all-ones B fragment, riding the same
// rescale as the O accumulator.
// ---------------------------------------------------------------------------
__global__ void __launch_bounds__(256)
attn_flash(const f16* __restrict__ Q, const f16* __restrict__ Kx,
           const f16* __restrict__ V, f16* __restrict__ O) {
    __shared__ f16 ks[32 * 64];       // [key][d]
    __shared__ f16 vsT[64 * 32];      // [d][key]  (transposed V)
    __shared__ f16 ps[8][16 * 32];    // per-wave P relay (C-layout -> A-layout)

    const int tid  = threadIdx.x;
    const int lane = tid & 31;
    const int wave = tid >> 5;
    const int bh   = blockIdx.y;
    const int b    = bh >> 4;
    const int h    = bh & 15;
    const int q0   = blockIdx.x * 128 + wave * 16;
    const size_t head = (size_t)bh * SEQ * HDIM;

    v16h qa[2];
#pragma unroll
    for (int c = 0; c < 2; ++c)
        qa[c] = load_frag_kcontig(Q + head, q0, HDIM, c * 32, lane);

    v16h ones;
#pragma unroll
    for (int i = 0; i < 16; ++i) ((FragU*)&ones)->e[i] = (f16)1.0f;

    float m[8];
    v8f   o[4], lacc;
#pragma unroll
    for (int r = 0; r < 8; ++r) m[r] = -1e30f;
#pragma unroll
    for (int dt = 0; dt < 4; ++dt) o[dt] = zero8();
    lacc = zero8();

    const float sc = 0.125f;  // 1/sqrt(HDIM)
    const int kr = (tid * 8) >> 6;   // staging: key row
    const int kd = (tid * 8) & 63;   // staging: d column base

    for (int t = 0; t < SEQ / 32; ++t) {
        __syncthreads();
        {
            const size_t g = head + (size_t)(t * 32) * HDIM + tid * 8;
#if USE_ASYNC
            async_cp16(ks + tid * 8, Kx + g);
#else
            *(uint4*)(ks + tid * 8) = *(const uint4*)(Kx + g);
#endif
            // V transposed into LDS: 8 scalar stores per thread, once per tile
            FragU vr;
            vr.q[0] = *(const uint4*)(V + g);
#pragma unroll
            for (int j = 0; j < 8; ++j) vsT[(kd + j) * 32 + kr] = vr.e[j];
        }
#if USE_ASYNC
        async_wait0();
#endif
        __syncthreads();

        // scores: S[16q x 32k] as two 16x16 accumulators
        v8f s0 = zero8(), s1 = zero8();
#pragma unroll
        for (int c = 0; c < 2; ++c) {
            v16h kb0 = load_frag_kcontig(ks, 0, HDIM, c * 32, lane);
            v16h kb1 = load_frag_kcontig(ks, 16, HDIM, c * 32, lane);
            s0 = wmma_f16(qa[c], kb0, s0);
            s1 = wmma_f16(qa[c], kb1, s1);
        }

        // online softmax: DPP row-max, exp, rescale accumulators, relay P
        const int prow = ((lane >> 4) << 3);
        const int pcol = lane & 15;
#pragma unroll
        for (int r = 0; r < 8; ++r) {
            const float a0 = s0[r] * sc, a1 = s1[r] * sc;
            const float mt = rowmax16(fmaxf(a0, a1));
            const float mn  = fmaxf(m[r], mt);
            const float rsc = __expf(m[r] - mn);
            const float e0  = __expf(a0 - mn);
            const float e1  = __expf(a1 - mn);
            m[r] = mn;
            lacc[r] *= rsc;
#pragma unroll
            for (int dt = 0; dt < 4; ++dt) o[dt][r] *= rsc;
            ps[wave][(r + prow) * 32 + pcol]      = (f16)e0;
            ps[wave][(r + prow) * 32 + pcol + 16] = (f16)e1;
        }
        // LDS is in-order per wave; ps is wave-private -> no barrier needed.
        v16h pa = load_frag_kcontig(&ps[wave][0], 0, 32, 0, lane);

        // O += P x V  (B fragments contiguous thanks to transposed staging)
#pragma unroll
        for (int dt = 0; dt < 4; ++dt) {
            v16h vb = load_frag_kcontig(vsT, dt * 16, 32, 0, lane);
            o[dt] = wmma_f16(pa, vb, o[dt]);
        }
        // row sums: l += P x ones (all 16 columns hold the row sum)
        lacc = wmma_f16(pa, ones, lacc);
    }

    // epilogue: normalize and write f16 into attn buffer [M, E]
#pragma unroll
    for (int r = 0; r < 8; ++r) {
        const float inv = 1.0f / lacc[r];
        const int row   = q0 + r + ((lane >> 4) << 3);
        const size_t mrow = (size_t)b * SEQ + row;
#pragma unroll
        for (int dt = 0; dt < 4; ++dt) {
            const int e = h * HDIM + dt * 16 + (lane & 15);
            O[mrow * EMBED + e] = (f16)(o[dt][r] * inv);
        }
    }
}

// ---------------------------------------------------------------------------
// Launch
// ---------------------------------------------------------------------------
extern "C" void kernel_launch(void* const* d_in, const int* in_sizes, int n_in,
                              void* d_out, int out_size, void* d_ws, size_t ws_size,
                              hipStream_t stream) {
    const float* x     = (const float*)d_in[0];
    const float* w_q   = (const float*)d_in[1];
    const float* b_q   = (const float*)d_in[2];
    const float* w_k   = (const float*)d_in[3];
    const float* b_k   = (const float*)d_in[4];
    const float* w_v   = (const float*)d_in[5];
    const float* b_v   = (const float*)d_in[6];
    const float* w_out = (const float*)d_in[7];
    const float* b_out = (const float*)d_in[8];

    const size_t NX = (size_t)MTOT * EMBED;
    const size_t NW = (size_t)EMBED * EMBED;

    f16* ws  = (f16*)d_ws;
    f16* xh  = ws;            // [M, E]
    f16* wqh = xh  + NX;
    f16* wkh = wqh + NW;
    f16* wvh = wkh + NW;
    f16* woh = wvh + NW;
    f16* qb  = woh + NW;      // [B, H, S, D]
    f16* kb  = qb  + NX;
    f16* vb  = kb  + NX;
    f16* ah  = vb  + NX;      // [M, E] attention output (f16)

    cast_f32_f16<<<2048, 256, 0, stream>>>(x, xh, (int)NX);
    cast_f32_f16<<<512, 256, 0, stream>>>(w_q, wqh, (int)NW);
    cast_f32_f16<<<512, 256, 0, stream>>>(w_k, wkh, (int)NW);
    cast_f32_f16<<<512, 256, 0, stream>>>(w_v, wvh, (int)NW);
    cast_f32_f16<<<512, 256, 0, stream>>>(w_out, woh, (int)NW);

    dim3 gg(MTOT / 128, EMBED / 128);
    gemm_wmma<1><<<gg, 256, 0, stream>>>(xh, wqh, b_q, qb, MTOT, EMBED, EMBED);
    gemm_wmma<1><<<gg, 256, 0, stream>>>(xh, wkh, b_k, kb, MTOT, EMBED, EMBED);
    gemm_wmma<1><<<gg, 256, 0, stream>>>(xh, wvh, b_v, vb, MTOT, EMBED, EMBED);

    attn_flash<<<dim3(SEQ / 128, BATCH * NHEADS), 256, 0, stream>>>(qb, kb, vb, ah);

    gemm_wmma<2><<<gg, 256, 0, stream>>>(ah, woh, b_out, d_out, MTOT, EMBED, EMBED);
}
